// GPTLanguageMoD_38817914421885
// MI455X (gfx1250) — matrix operand order repaired and checked by reading, not compiled, not run on hardware
//
#include <hip/hip_runtime.h>

// ---------------- problem dims (fixed by the reference) ----------------
#define B_   4
#define T_   1024
#define V_   50257
#define C_   768
#define H_   6
#define HS_  128
#define FF_  3072
#define L_   6
#define BT_  (B_ * T_)

// ---------------- WMMA types & helpers (gfx1250, wave32) ----------------
typedef __attribute__((ext_vector_type(16))) __bf16 v16bf;
typedef __attribute__((ext_vector_type(8)))  __bf16 v8bf;
typedef __attribute__((ext_vector_type(8)))  float  v8f;

__device__ __forceinline__ unsigned short f2bf(float f) {
  union { float f; unsigned u; } a; a.f = f;
  unsigned u = a.u;
  u += 0x7fffu + ((u >> 16) & 1u);          // round-to-nearest-even
  return (unsigned short)(u >> 16);
}

__device__ __forceinline__ v8f vzero8() {
  v8f z;
#pragma unroll
  for (int i = 0; i < 8; ++i) z[i] = 0.0f;
  return z;
}

// A-operand fragment: 16(M) x 32(K) bf16, LDS row-major [m][k].
// Lanes 0-15 -> row m=lane, K {0..7,16..23}; lanes 16-31 -> K {8..15,24..31}.
__device__ __forceinline__ v16bf fragA(const unsigned short* base, int pitch) {
  const int lane = threadIdx.x & 31;
  const unsigned short* p = base + (lane & 15) * pitch + ((lane & 16) ? 8 : 0);
  v8bf lo = *(const v8bf*)(p);
  v8bf hi = *(const v8bf*)(p + 16);
  v16bf o;
#pragma unroll
  for (int i = 0; i < 8; ++i) { o[i] = lo[i]; o[8 + i] = hi[i]; }
  return o;
}

// B-operand fragment: 32(K) x 16(N), stored transposed in LDS as [n][k].
// Lanes 0-15 -> col n=lane, K {0..7,16..23}; lanes 16-31 -> K {8..15,24..31}.
__device__ __forceinline__ v16bf fragB(const unsigned short* base, int pitch) {
  const int lane = threadIdx.x & 31;
  const unsigned short* p = base + (lane & 15) * pitch + ((lane & 16) ? 8 : 0);
  v8bf lo = *(const v8bf*)(p);
  v8bf hi = *(const v8bf*)(p + 16);
  v16bf o;
#pragma unroll
  for (int i = 0; i < 8; ++i) { o[i] = lo[i]; o[8 + i] = hi[i]; }
  return o;
}

__device__ __forceinline__ v8f wmma_bf16(v16bf a, v16bf b, v8f c) {
  return __builtin_amdgcn_wmma_f32_16x16x32_bf16(false, a, false, b, (short)0, c, false, false);
}

// ---------------- one-time weight pack: fp32 [K][N] (opt. head-blocked) -> bf16 [N][K] ----------------
__global__ __launch_bounds__(256) void k_packT(
    const float* __restrict__ src, int ldb, int head_ncols, long long head_stride,
    long long src_lstride, unsigned short* __restrict__ dst, long long dst_lstride,
    int K, int N) {
  __shared__ float t[32][33];
  const float* s = src + (long long)blockIdx.z * src_lstride;
  unsigned short* d = dst + (long long)blockIdx.z * dst_lstride;
  const int n0 = blockIdx.x * 32, k0 = blockIdx.y * 32;
  const int tx = threadIdx.x, ty = threadIdx.y;
  {
    const long long n = n0 + tx;
    const bool nok = n < N;
    long long hbase = 0;
    if (nok) hbase = (n / head_ncols) * head_stride + (n % head_ncols);
#pragma unroll
    for (int p = 0; p < 4; ++p) {
      const int k = k0 + ty + p * 8;
      t[ty + p * 8][tx] = nok ? s[hbase + (long long)k * ldb] : 0.0f;
    }
  }
  __syncthreads();
#pragma unroll
  for (int p = 0; p < 4; ++p) {
    const long long n = n0 + ty + p * 8;
    if (n < N) d[n * K + k0 + tx] = f2bf(t[tx][ty + p * 8]);
  }
}

// ---------------- embeddings (x stays fp32: residual stream) ----------------
__global__ __launch_bounds__(256) void k_embed(const int* __restrict__ idx,
    const float* __restrict__ tok, const float* __restrict__ pos, float* __restrict__ x) {
  const int t = blockIdx.x;
  const long long trow = (long long)idx[t] * C_;
  const int prow = (t % T_) * C_;
  for (int c = threadIdx.x; c < C_; c += 256)
    x[t * C_ + c] = tok[trow + c] + pos[prow + c];
}

// ---------------- MoD router ----------------
__global__ __launch_bounds__(256) void k_router(const float* __restrict__ x,
    const float* __restrict__ wr, const float* __restrict__ br,
    const float* __restrict__ wa, const float* __restrict__ ba,
    float* __restrict__ rw, int* __restrict__ sel) {
  __shared__ float r1[256], r2[256];
  const int t = blockIdx.x;
  const float* xr = x + t * C_;
  float s1 = 0.0f, s2 = 0.0f;
  for (int c = threadIdx.x; c < C_; c += 256) {
    const float xv = xr[c];
    s1 += xv * wr[c];
    s2 += xv * wa[c];
  }
  r1[threadIdx.x] = s1; r2[threadIdx.x] = s2;
  __syncthreads();
  for (int o = 128; o > 0; o >>= 1) {
    if (threadIdx.x < o) { r1[threadIdx.x] += r1[threadIdx.x + o]; r2[threadIdx.x] += r2[threadIdx.x + o]; }
    __syncthreads();
  }
  if (threadIdx.x == 0) {
    rw[t]  = r1[0] + br[0];
    sel[t] = (r2[0] + ba[0]) > 0.0f ? 1 : 0;
  }
}

// ---------------- LayerNorm: fp32 in, bf16 out ----------------
__global__ __launch_bounds__(256) void k_ln(const float* __restrict__ xin,
    const float* __restrict__ g, const float* __restrict__ bb,
    unsigned short* __restrict__ o) {
  __shared__ float rs[256], rss[256];
  const int t = blockIdx.x;
  const float* xr = xin + t * C_;
  float s = 0.0f, ss = 0.0f;
  for (int c = threadIdx.x; c < C_; c += 256) { const float v = xr[c]; s += v; ss += v * v; }
  rs[threadIdx.x] = s; rss[threadIdx.x] = ss;
  __syncthreads();
  for (int oo = 128; oo > 0; oo >>= 1) {
    if (threadIdx.x < oo) { rs[threadIdx.x] += rs[threadIdx.x + oo]; rss[threadIdx.x] += rss[threadIdx.x + oo]; }
    __syncthreads();
  }
  const float mean = rs[0]  * (1.0f / C_);
  const float var  = rss[0] * (1.0f / C_) - mean * mean;
  const float inv  = rsqrtf(var + 1e-5f);
  for (int c = threadIdx.x; c < C_; c += 256)
    o[t * C_ + c] = f2bf((xr[c] - mean) * inv * g[c] + bb[c]);
}

// ---------------- bf16-WMMA GEMM with register-staged pipeline ----------------
// A bf16 [M,K], Bt bf16 [N][K] pre-transposed; fp32 accum; out fp32 or bf16.
// Tile kt+1 global loads are issued before tile kt's WMMAs so load latency is
// hidden behind compute; ds_store of the registers happens next iteration.
// All hot-path offsets fit int32 for this problem (max 4096*50257 < 2^31).
__global__ __launch_bounds__(256) void k_gemm(
    const unsigned short* __restrict__ A, int lda,
    const unsigned short* __restrict__ Bt,
    const float* __restrict__ bias, const float* __restrict__ resid,
    void* __restrict__ Co, int ldc, int M, int N, int K, int relu, int out_bf16) {
  __shared__ __attribute__((aligned(16))) unsigned short As[128 * 32];
  __shared__ __attribute__((aligned(16))) unsigned short Bs[128 * 32];  // [n][k]

  const int tid  = threadIdx.x;
  const int lane = tid & 31;
  const int wave = tid >> 5;
  const int wm   = wave & 3;     // 4 M-strips of 32 rows
  const int wn   = wave >> 2;    // 2 N-strips of 64 cols
  const int m0 = blockIdx.y * 128;
  const int n0 = blockIdx.x * 128;
  const bool edge = (n0 + 128 > N);   // block-uniform: only LM-head edge blocks

  v8f acc[2][4];
#pragma unroll
  for (int i = 0; i < 2; ++i)
#pragma unroll
    for (int j = 0; j < 4; ++j) acc[i][j] = vzero8();

  // staging coords: each thread covers rows r0 and r0+64, one 8-elem chunk
  const int r0 = tid >> 2;            // 0..63
  const int c8 = (tid & 3) * 8;       // 0,8,16,24
  const unsigned short* Arow0 = A  + (m0 + r0)      * lda + c8;
  const unsigned short* Arow1 = A  + (m0 + r0 + 64) * lda + c8;
  const unsigned short* Brow0 = Bt + (n0 + r0)      * K   + c8;
  const unsigned short* Brow1 = Bt + (n0 + r0 + 64) * K   + c8;
  const bool bok0 = (n0 + r0)      < N;
  const bool bok1 = (n0 + r0 + 64) < N;
  const uint4 zz = make_uint4(0u, 0u, 0u, 0u);

  // preload tile 0 into registers
  uint4 ra0 = *(const uint4*)(Arow0);
  uint4 ra1 = *(const uint4*)(Arow1);
  uint4 rb0, rb1;
  if (edge) {
    rb0 = bok0 ? *(const uint4*)(Brow0) : zz;
    rb1 = bok1 ? *(const uint4*)(Brow1) : zz;
  } else {
    rb0 = *(const uint4*)(Brow0);
    rb1 = *(const uint4*)(Brow1);
  }

  const int nk = K / 32;
  int kof = 0;
  for (int kt = 0; kt < nk; ++kt) {
    __syncthreads();  // previous compute finished reading LDS
    *(uint4*)&As[r0 * 32 + c8]        = ra0;
    *(uint4*)&As[(r0 + 64) * 32 + c8] = ra1;
    *(uint4*)&Bs[r0 * 32 + c8]        = rb0;
    *(uint4*)&Bs[(r0 + 64) * 32 + c8] = rb1;
    __syncthreads();  // tile kt visible
    if (kt + 1 < nk) {  // issue tile kt+1 loads; they retire behind the WMMAs
      kof += 32;
      ra0 = *(const uint4*)(Arow0 + kof);
      ra1 = *(const uint4*)(Arow1 + kof);
      if (edge) {
        rb0 = bok0 ? *(const uint4*)(Brow0 + kof) : zz;
        rb1 = bok1 ? *(const uint4*)(Brow1 + kof) : zz;
      } else {
        rb0 = *(const uint4*)(Brow0 + kof);
        rb1 = *(const uint4*)(Brow1 + kof);
      }
    }
    v16bf af[2], bfm[4];
#pragma unroll
    for (int mt = 0; mt < 2; ++mt) af[mt]  = fragA(&As[(wm * 32 + mt * 16) * 32], 32);
#pragma unroll
    for (int nt = 0; nt < 4; ++nt) bfm[nt] = fragB(&Bs[(wn * 64 + nt * 16) * 32], 32);
#pragma unroll
    for (int mt = 0; mt < 2; ++mt)
#pragma unroll
      for (int nt = 0; nt < 4; ++nt)
        acc[mt][nt] = wmma_bf16(af[mt], bfm[nt], acc[mt][nt]);
  }

  // epilogue: D layout -> lane col = lane&15, rows = i + (lane>=16 ? 8 : 0)
  const int colb = lane & 15;
  const int rowh = (lane & 16) ? 8 : 0;
#pragma unroll
  for (int mt = 0; mt < 2; ++mt) {
#pragma unroll
    for (int nt = 0; nt < 4; ++nt) {
      const int col = n0 + wn * 64 + nt * 16 + colb;
      if (col >= N) continue;
      const float bia = bias ? bias[col] : 0.0f;
#pragma unroll
      for (int i = 0; i < 8; ++i) {
        const int row = m0 + wm * 32 + mt * 16 + rowh + i;
        float v = acc[mt][nt][i] + bia;
        if (relu) v = fmaxf(v, 0.0f);
        if (resid) v += resid[row * ldc + col];
        if (out_bf16) ((unsigned short*)Co)[row * ldc + col] = f2bf(v);
        else          ((float*)Co)[row * ldc + col] = v;
      }
    }
  }
}

// ---------------- flash attention with MoD + causal masking (bf16 q/k/v/att) ----------------
__global__ __launch_bounds__(32) void k_attn(
    const unsigned short* __restrict__ qm, const unsigned short* __restrict__ km,
    const unsigned short* __restrict__ vm, const int* __restrict__ sel,
    unsigned short* __restrict__ att) {
  __shared__ __attribute__((aligned(16))) unsigned short Qs[16 * 128];
  __shared__ __attribute__((aligned(16))) unsigned short Ks[32 * 128];
  __shared__ __attribute__((aligned(16))) unsigned short Vt[128 * 32];  // [d][key]
  __shared__ __attribute__((aligned(16))) unsigned short Ps[16 * 32];   // [q][key]

  const int lane = threadIdx.x;
  const int q0 = blockIdx.x * 16;
  const int h  = blockIdx.y;
  const int b  = blockIdx.z;
  const float scale = 0.08838834764831845f;  // 128^-0.5

  {  // stage Q tile (16 x 128): straight b128 copies
    const int r  = lane & 15;
    const int ch = (lane >> 4) * 64;
    const uint4* s = (const uint4*)(qm + (b * T_ + q0 + r) * C_ + h * HS_ + ch);
    uint4* d = (uint4*)&Qs[r * 128 + ch];
#pragma unroll
    for (int i = 0; i < 8; ++i) d[i] = s[i];
  }

  const int qg   = q0 + (lane & 15);
  const int selq = sel[b * T_ + qg];
  const int krow = (lane & 16) ? 8 : 0;

  float mrun = -__builtin_inff();
  float srun = 0.0f;
  v8f accd[8];
#pragma unroll
  for (int dt = 0; dt < 8; ++dt) accd[dt] = vzero8();

  const int nkb = (q0 + 16 + 31) >> 5;  // causal block bound
  for (int kb = 0; kb < nkb; ++kb) {
    const int kbase = kb * 32;
    const int selk  = sel[b * T_ + kbase + lane];
    const unsigned kmask = (unsigned)__ballot(selk != 0);

    {  // stage K block (32 x 128), one row per lane, b128 copies
      const uint4* s = (const uint4*)(km + (b * T_ + kbase + lane) * C_ + h * HS_);
      uint4* d = (uint4*)&Ks[lane * 128];
#pragma unroll
      for (int i = 0; i < 16; ++i) d[i] = s[i];
    }
    {  // stage V block transposed: Vt[d][key]
      const uint4* s = (const uint4*)(vm + (b * T_ + kbase + lane) * C_ + h * HS_);
#pragma unroll
      for (int c8 = 0; c8 < 16; ++c8) {
        const uint4 u = s[c8];
        const int d0 = c8 * 8;
        Vt[(d0 + 0) * 32 + lane] = (unsigned short)(u.x);
        Vt[(d0 + 1) * 32 + lane] = (unsigned short)(u.x >> 16);
        Vt[(d0 + 2) * 32 + lane] = (unsigned short)(u.y);
        Vt[(d0 + 3) * 32 + lane] = (unsigned short)(u.y >> 16);
        Vt[(d0 + 4) * 32 + lane] = (unsigned short)(u.z);
        Vt[(d0 + 5) * 32 + lane] = (unsigned short)(u.z >> 16);
        Vt[(d0 + 6) * 32 + lane] = (unsigned short)(u.w);
        Vt[(d0 + 7) * 32 + lane] = (unsigned short)(u.w >> 16);
      }
    }
    __syncthreads();  // single-wave WG -> S_NOP

    // scores^T: D[key, q] = K_block x Q^T
    v8f s0 = vzero8(), s1 = vzero8();
#pragma unroll
    for (int c0 = 0; c0 < 128; c0 += 32) {
      const v16bf bq = fragB(&Qs[c0], 128);
      const v16bf a0 = fragA(&Ks[0 * 128 + c0], 128);
      const v16bf a1 = fragA(&Ks[16 * 128 + c0], 128);
      s0 = wmma_bf16(a0, bq, s0);
      s1 = wmma_bf16(a1, bq, s1);
    }

    // mask + online softmax (lane column q = lane&15; rows = keys)
    float p0[8], p1[8];
    float mloc = -__builtin_inff();
#pragma unroll
    for (int i = 0; i < 8; ++i) {
      const int kl0 = i + krow, kl1 = 16 + i + krow;
      const bool ok0 = selq && ((kmask >> kl0) & 1u) && (kbase + kl0 <= qg);
      const bool ok1 = selq && ((kmask >> kl1) & 1u) && (kbase + kl1 <= qg);
      p0[i] = ok0 ? s0[i] * scale : -1e30f;
      p1[i] = ok1 ? s1[i] * scale : -1e30f;
      mloc = fmaxf(mloc, fmaxf(p0[i], p1[i]));
    }
    mloc = fmaxf(mloc, __shfl_xor(mloc, 16));
    const float mnew = fmaxf(mrun, mloc);
    const float rsc  = __expf(mrun - mnew);
    float lsum = 0.0f;
#pragma unroll
    for (int i = 0; i < 8; ++i) {
      p0[i] = __expf(p0[i] - mnew);
      p1[i] = __expf(p1[i] - mnew);
      lsum += p0[i] + p1[i];
    }
    lsum += __shfl_xor(lsum, 16);
    srun = srun * rsc + lsum;
    mrun = mnew;

    {  // store P (16 q x 32 keys)
      const int qc = lane & 15;
#pragma unroll
      for (int i = 0; i < 8; ++i) {
        Ps[qc * 32 + (i + krow)]      = f2bf(p0[i]);
        Ps[qc * 32 + 16 + (i + krow)] = f2bf(p1[i]);
      }
    }
    // rescale accumulators (per-query-row factors, gathered cross-lane)
    float rr[8];
#pragma unroll
    for (int i = 0; i < 8; ++i) rr[i] = __shfl(rsc, i + krow);
#pragma unroll
    for (int dt = 0; dt < 8; ++dt)
#pragma unroll
      for (int i = 0; i < 8; ++i) accd[dt][i] *= rr[i];
    __syncthreads();

    // att += P x V
    const v16bf ap = fragA(&Ps[0], 32);
#pragma unroll
    for (int dt = 0; dt < 8; ++dt) {
      const v16bf bv = fragB(&Vt[dt * 16 * 32], 32);
      accd[dt] = wmma_bf16(ap, bv, accd[dt]);
    }
    __syncthreads();
  }

  float sinv[8];
#pragma unroll
  for (int i = 0; i < 8; ++i) sinv[i] = 1.0f / __shfl(srun, i + krow);
#pragma unroll
  for (int dt = 0; dt < 8; ++dt)
#pragma unroll
    for (int i = 0; i < 8; ++i) {
      const int row = b * T_ + q0 + i + krow;
      att[row * C_ + h * HS_ + dt * 16 + (lane & 15)] = f2bf(accd[dt][i] * sinv[i]);
    }
}

// ---------------- MoD scatter: x = sel ? blk * rw : x ----------------
__global__ __launch_bounds__(256) void k_scatter(float* __restrict__ x,
    const float* __restrict__ blk, const float* __restrict__ rw, const int* __restrict__ sel) {
  const int t = blockIdx.x;
  if (!sel[t]) return;
  const float w = rw[t];
  for (int c = threadIdx.x; c < C_; c += 256)
    x[t * C_ + c] = blk[t * C_ + c] * w;
}

// ---------------- host orchestration ----------------
extern "C" void kernel_launch(void* const* d_in, const int* in_sizes, int n_in,
                              void* d_out, int out_size, void* d_ws, size_t ws_size,
                              hipStream_t stream) {
  (void)in_sizes; (void)n_in; (void)out_size; (void)ws_size;
  const int*   idx      = (const int*)  d_in[0];
  const float* tok_emb  = (const float*)d_in[1];
  const float* pos_emb  = (const float*)d_in[2];
  const float* router_w = (const float*)d_in[3];
  const float* router_b = (const float*)d_in[4];
  const float* aux_w    = (const float*)d_in[5];
  const float* aux_b    = (const float*)d_in[6];
  const float* ln1_g    = (const float*)d_in[7];
  const float* ln1_b    = (const float*)d_in[8];
  const float* ln2_g    = (const float*)d_in[9];
  const float* ln2_b    = (const float*)d_in[10];
  const float* wq       = (const float*)d_in[11];
  const float* wk       = (const float*)d_in[12];
  const float* wv       = (const float*)d_in[13];
  const float* proj_w   = (const float*)d_in[14];
  const float* proj_b   = (const float*)d_in[15];
  const float* ffn_w1   = (const float*)d_in[16];
  const float* ffn_b1   = (const float*)d_in[17];
  const float* ffn_w2   = (const float*)d_in[18];
  const float* ffn_b2   = (const float*)d_in[19];
  const float* lnf_g    = (const float*)d_in[20];
  const float* lnf_b    = (const float*)d_in[21];
  const float* lm_w     = (const float*)d_in[22];
  const float* lm_b     = (const float*)d_in[23];
  float* out = (float*)d_out;

  // ---- workspace layout (256B-aligned carve-outs) ----
  size_t off = 0;
  char* base = (char*)d_ws;
  auto carve = [&](size_t bytes) -> void* {
    void* p = base + off;
    off += (bytes + 255) & ~(size_t)255;
    return p;
  };
  const long long XE = (long long)BT_ * C_;
  float*          x    = (float*)carve(XE * 4);
  float*          yb   = (float*)carve(XE * 4);
  float*          rwb  = (float*)carve(BT_ * 4);
  int*            sel  = (int*)  carve(BT_ * 4);
  unsigned short* hbB  = (unsigned short*)carve(XE * 2);
  unsigned short* qB   = (unsigned short*)carve(XE * 2);   // blk aliases q/k region
  unsigned short* kB   = (unsigned short*)carve(XE * 2);
  unsigned short* vB   = (unsigned short*)carve(XE * 2);
  unsigned short* abB  = (unsigned short*)carve(XE * 2);
  unsigned short* ffB  = (unsigned short*)carve((long long)BT_ * FF_ * 2);
  const long long WCC = (long long)C_ * C_;
  unsigned short* WqT  = (unsigned short*)carve((long long)L_ * WCC * 2);
  unsigned short* WkT  = (unsigned short*)carve((long long)L_ * WCC * 2);
  unsigned short* WvT  = (unsigned short*)carve((long long)L_ * WCC * 2);
  unsigned short* WpT  = (unsigned short*)carve((long long)L_ * WCC * 2);
  unsigned short* W1T  = (unsigned short*)carve((long long)L_ * C_ * FF_ * 2);
  unsigned short* W2T  = (unsigned short*)carve((long long)L_ * FF_ * C_ * 2);
  unsigned short* WlmT = (unsigned short*)carve((long long)V_ * C_ * 2);
  float* blk = (float*)qB;  // fp32 BT*C fits in q+k bf16 region; both dead when used

  const dim3 tb256(256), tbT(32, 8);
  const long long hs = (long long)C_ * HS_;

  // ---- one-time (per launch) weight packs: fp32 -> bf16, transposed [N][K] ----
  k_packT<<<dim3(C_ / 32, C_ / 32, L_), tbT, 0, stream>>>(wq, HS_, HS_, hs, (long long)H_ * hs, WqT, WCC, C_, C_);
  k_packT<<<dim3(C_ / 32, C_ / 32, L_), tbT, 0, stream>>>(wk, HS_, HS_, hs, (long long)H_ * hs, WkT, WCC, C_, C_);
  k_packT<<<dim3(C_ / 32, C_ / 32, L_), tbT, 0, stream>>>(wv, HS_, HS_, hs, (long long)H_ * hs, WvT, WCC, C_, C_);
  k_packT<<<dim3(C_ / 32, C_ / 32, L_), tbT, 0, stream>>>(proj_w, C_, C_, 0, WCC, WpT, WCC, C_, C_);
  k_packT<<<dim3(FF_ / 32, C_ / 32, L_), tbT, 0, stream>>>(ffn_w1, FF_, FF_, 0, (long long)C_ * FF_, W1T, (long long)C_ * FF_, C_, FF_);
  k_packT<<<dim3(C_ / 32, FF_ / 32, L_), tbT, 0, stream>>>(ffn_w2, C_, C_, 0, (long long)FF_ * C_, W2T, (long long)FF_ * C_, FF_, C_);
  k_packT<<<dim3((V_ + 31) / 32, C_ / 32, 1), tbT, 0, stream>>>(lm_w, V_, V_, 0, 0, WlmT, 0, C_, V_);

  const dim3 gC(C_ / 128, BT_ / 128);            // 6 x 32
  const dim3 gF(FF_ / 128, BT_ / 128);           // 24 x 32
  const dim3 gV((V_ + 127) / 128, BT_ / 128);    // 393 x 32
  const dim3 gAtt(T_ / 16, H_, B_);              // 1 wave each

  k_embed<<<BT_, tb256, 0, stream>>>(idx, tok_emb, pos_emb, x);

  for (int l = 0; l < L_; ++l) {
    k_router<<<BT_, tb256, 0, stream>>>(x, router_w + (long long)l * C_, router_b + l,
                                        aux_w + (long long)l * C_, aux_b + l, rwb, sel);
    k_ln<<<BT_, tb256, 0, stream>>>(x, ln1_g + (long long)l * C_, ln1_b + (long long)l * C_, hbB);
    k_gemm<<<gC, tb256, 0, stream>>>(hbB, C_, WqT + (long long)l * WCC, nullptr, nullptr,
                                     qB, C_, BT_, C_, C_, 0, 1);
    k_gemm<<<gC, tb256, 0, stream>>>(hbB, C_, WkT + (long long)l * WCC, nullptr, nullptr,
                                     kB, C_, BT_, C_, C_, 0, 1);
    k_gemm<<<gC, tb256, 0, stream>>>(hbB, C_, WvT + (long long)l * WCC, nullptr, nullptr,
                                     vB, C_, BT_, C_, C_, 0, 1);
    k_attn<<<gAtt, dim3(32), 0, stream>>>(qB, kB, vB, sel, abB);
    k_gemm<<<gC, tb256, 0, stream>>>(abB, C_, WpT + (long long)l * WCC,
                                     proj_b + (long long)l * C_, x, yb, C_, BT_, C_, C_, 0, 0);
    k_ln<<<BT_, tb256, 0, stream>>>(yb, ln2_g + (long long)l * C_, ln2_b + (long long)l * C_, hbB);
    k_gemm<<<gF, tb256, 0, stream>>>(hbB, C_, W1T + (long long)l * C_ * FF_,
                                     ffn_b1 + (long long)l * FF_, nullptr, ffB, FF_, BT_, FF_, C_, 1, 1);
    k_gemm<<<gC, tb256, 0, stream>>>(ffB, FF_, W2T + (long long)l * FF_ * C_,
                                     ffn_b2 + (long long)l * C_, yb, blk, C_, BT_, C_, FF_, 0, 0);
    k_scatter<<<BT_, tb256, 0, stream>>>(x, blk, rwb, sel);
  }
  k_ln<<<BT_, tb256, 0, stream>>>(x, lnf_g, lnf_b, hbB);
  k_gemm<<<gV, tb256, 0, stream>>>(hbB, C_, WlmT, lm_b, nullptr, out, V_, BT_, V_, C_, 0, 0);
}
